// SpatialAttention_25202868093136
// MI455X (gfx1250) — matrix operand order repaired and checked by reading, not compiled, not run on hardware
//
#include <hip/hip_runtime.h>

typedef __bf16 bh_t;
typedef __attribute__((ext_vector_type(16))) __bf16 v16bf;
typedef __attribute__((ext_vector_type(8)))  __bf16 v8bf;
typedef __attribute__((ext_vector_type(8)))  float  v8f;

union V16U { v16bf v; v8bf h[2]; };

// A-operand chunk (16-bit, 16x32): lane holds row, halves 0-7 = K[koff8+0..7],
// halves 8-15 = K[koff8+16..23]; caller pre-adds koff8 = 8*(lane>=16).
static __device__ __forceinline__ v16bf loadA(const bh_t* p) {
  V16U u;
  u.h[0] = *(const v8bf*)(p);
  u.h[1] = *(const v8bf*)(p + 16);
  return u.v;
}
// B-operand chunk (16-bit, 32x16): lane holds column, halves 0-7 = K[koff8+0..7],
// halves 8-15 = K[koff8+16..23]; caller pre-adds koff8 = 8*(lane>=16).
static __device__ __forceinline__ v16bf loadB(const bh_t* p) {
  V16U u;
  u.h[0] = *(const v8bf*)(p);
  u.h[1] = *(const v8bf*)(p + 16);
  return u.v;
}

#define WMMA_BF16(A,B,C) \
  __builtin_amdgcn_wmma_f32_16x16x32_bf16(false,(A),false,(B),(short)0,(C),false,false)

// -------- constants --------
// B=32 batches, C=512 channels, D=64 qk-dim, N=1024 spatial

// ---------------------------------------------------------------------------
// Kernel 0a: xpT[b][n][c] = bf16( x[b][c][n] + pos[c][n] )
// ---------------------------------------------------------------------------
__global__ void k_prep_xpT(const float* __restrict__ x, const float* __restrict__ pos,
                           bh_t* __restrict__ xpT) {
  const int total = 32 * 1024 * 512;
  for (int idx = blockIdx.x * blockDim.x + threadIdx.x; idx < total;
       idx += gridDim.x * blockDim.x) {
    int c = idx & 511;
    int n = (idx >> 9) & 1023;
    int b = idx >> 19;
    float v = x[((size_t)(b * 512 + c)) * 1024 + n] + pos[c * 1024 + n];
    xpT[idx] = (bh_t)v;
  }
}

// ---------------------------------------------------------------------------
// Kernel 0b: f32 -> bf16 weight conversion
// ---------------------------------------------------------------------------
__global__ void k_cvt_bf16(const float* __restrict__ src, bh_t* __restrict__ dst, int n) {
  for (int i = blockIdx.x * blockDim.x + threadIdx.x; i < n; i += gridDim.x * blockDim.x)
    dst[i] = (bh_t)src[i];
}

// ---------------------------------------------------------------------------
// Kernel 1: q/k/v projections. One wave per 16x16 output tile, K=512.
// m-tiles: 0..3 -> q, 4..7 -> k, 8..39 -> v   (40 m-tiles x 64 n-tiles x 32 b)
// q,k stored transposed [b][n][64]; v stored [b][c][n].
// ---------------------------------------------------------------------------
__global__ __launch_bounds__(256) void k_qkv(
    const bh_t* __restrict__ xpT,
    const bh_t* __restrict__ qwb, const bh_t* __restrict__ kwb, const bh_t* __restrict__ vwb,
    const float* __restrict__ qb, const float* __restrict__ kb, const float* __restrict__ vb,
    bh_t* __restrict__ qT, bh_t* __restrict__ kT, bh_t* __restrict__ vR) {
  const int lane = threadIdx.x & 31, wave = threadIdx.x >> 5;
  const int lo = lane & 15, hi = lane >> 4;
  const int tile = blockIdx.x * 8 + wave;
  const int b = tile / 2560;
  const int r = tile % 2560;
  const int mt = r >> 6;
  const int nt = r & 63;

  const bh_t* W;
  const float* bias;
  if (mt < 4)      { W = qwb + (size_t)mt * 16 * 512;       bias = qb + mt * 16; }
  else if (mt < 8) { W = kwb + (size_t)(mt - 4) * 16 * 512; bias = kb + (mt - 4) * 16; }
  else             { W = vwb + (size_t)(mt - 8) * 16 * 512; bias = vb + (mt - 8) * 16; }

  const int coln = nt * 16 + lo;
  const bh_t* arow = W + lo * 512 + hi * 8;                        // A: weight row
  const bh_t* bcol = xpT + ((size_t)(b * 1024 + coln)) * 512 + hi * 8;  // B: xp column

  v8f acc = (v8f)0.0f;
#pragma unroll
  for (int kc = 0; kc < 512; kc += 32) {
    v16bf a  = loadA(arow + kc);
    v16bf bb = loadB(bcol + kc);
    acc = WMMA_BF16(a, bb, acc);
  }

  if (mt < 8) {
    bh_t* dst = (mt < 4) ? qT : kT;
    const int d0 = (mt & 3) * 16 + hi * 8;
    v8bf o;
#pragma unroll
    for (int v = 0; v < 8; v++) o[v] = (bh_t)(acc[v] + bias[v + hi * 8]);
    *(v8bf*)(dst + ((size_t)(b * 1024 + coln)) * 64 + d0) = o;
  } else {
    const int c0 = (mt - 8) * 16 + hi * 8;
#pragma unroll
    for (int v = 0; v < 8; v++)
      vR[((size_t)(b * 512 + c0 + v)) * 1024 + coln] = (bh_t)(acc[v] + bias[v + hi * 8]);
  }
}

// ---------------------------------------------------------------------------
// Kernel 2: fused flash attention + output projection.
// grid = (64 i-tiles, 32 batches), 8 waves. Wave 0: scores + online softmax
// producer; all 8 waves: PV consumers over their 64-channel slice; then the
// 512x512 out-projection is done from LDS in the same workgroup.
// ---------------------------------------------------------------------------
__global__ __launch_bounds__(256) void k_attn(
    const bh_t* __restrict__ qT, const bh_t* __restrict__ kT, const bh_t* __restrict__ vR,
    const bh_t* __restrict__ owb, const float* __restrict__ ob,
    float* __restrict__ out) {
  __shared__ __align__(16) bh_t  p_lds[16 * 40];     // P block [i][j], stride 40
  __shared__ float scale_lds[16];                    // per-row rescale factor
  __shared__ float linv_lds[16];                     // per-row 1/l
  __shared__ __align__(16) bh_t  inter_lds[16 * 520]; // attn-out [i][c], stride 520

  const int lane = threadIdx.x & 31, wave = threadIdx.x >> 5;
  const int lo = lane & 15, hi = lane >> 4;
  const int it = blockIdx.x, b = blockIdx.y;
  const int i0 = it * 16;

  v16bf aq0, aq1;
  float m[8], l[8];
  if (wave == 0) {
    const bh_t* qrow = qT + ((size_t)(b * 1024 + i0 + lo)) * 64 + hi * 8;
    aq0 = loadA(qrow);        // d = 0..31
    aq1 = loadA(qrow + 32);   // d = 32..63
#pragma unroll
    for (int r = 0; r < 8; r++) { m[r] = -1e30f; l[r] = 0.0f; }
  }

  const int c0 = wave * 64;
  v8f acc[4];
#pragma unroll
  for (int t = 0; t < 4; t++) acc[t] = (v8f)0.0f;

  for (int step = 0; step < 32; step++) {
    const int j0 = step * 32;
    if (wave == 0) {
      // scores for columns j0..j0+31 (two 16x16 tiles)
      const bh_t* kc0 = kT + ((size_t)(b * 1024 + j0 + lo)) * 64 + hi * 8;
      const bh_t* kc1 = kT + ((size_t)(b * 1024 + j0 + 16 + lo)) * 64 + hi * 8;
      v8f s0 = (v8f)0.0f, s1 = (v8f)0.0f;
      s0 = WMMA_BF16(aq0, loadB(kc0),      s0);
      s0 = WMMA_BF16(aq1, loadB(kc0 + 32), s0);
      s1 = WMMA_BF16(aq0, loadB(kc1),      s1);
      s1 = WMMA_BF16(aq1, loadB(kc1 + 32), s1);

      // online softmax over this 32-column block; rows live across 16-lane halves
#pragma unroll
      for (int r = 0; r < 8; r++) {
        float cur = fmaxf(s0[r], s1[r]);
#pragma unroll
        for (int off = 1; off < 16; off <<= 1) cur = fmaxf(cur, __shfl_xor(cur, off, 32));
        const float mn  = fmaxf(m[r], cur);
        const float scl = __expf(m[r] - mn);
        const float p0  = __expf(s0[r] - mn);
        const float p1  = __expf(s1[r] - mn);
        float rs = p0 + p1;
#pragma unroll
        for (int off = 1; off < 16; off <<= 1) rs += __shfl_xor(rs, off, 32);
        l[r] = l[r] * scl + rs;
        m[r] = mn;
        if (lo == 0) scale_lds[r + hi * 8] = scl;
        const int row = r + hi * 8;
        p_lds[row * 40 + lo]      = (bh_t)p0;
        p_lds[row * 40 + lo + 16] = (bh_t)p1;
      }
    }
    __syncthreads();

    // consumers: acc <- acc*scale + V(c,j) x P^T(j,i)
    const float sc = scale_lds[lo];
    const v16bf pb = loadB(p_lds + lo * 40 + hi * 8);
#pragma unroll
    for (int t = 0; t < 4; t++) {
      const bh_t* vp = vR + ((size_t)(b * 512 + c0 + t * 16 + lo)) * 1024 + j0 + hi * 8;
      v16bf va = loadA(vp);
      acc[t] = acc[t] * sc;
      acc[t] = WMMA_BF16(va, pb, acc[t]);
    }
    __syncthreads();  // protect p_lds/scale_lds before next producer round
  }

  if (wave == 0 && lo == 0) {
#pragma unroll
    for (int r = 0; r < 8; r++) linv_lds[r + hi * 8] = 1.0f / l[r];
  }
  __syncthreads();

  // normalize and stage attn-out [i][c] in LDS (bf16)
  const float li = linv_lds[lo];
#pragma unroll
  for (int t = 0; t < 4; t++) {
    v8bf o;
#pragma unroll
    for (int v = 0; v < 8; v++) o[v] = (bh_t)(acc[t][v] * li);
    *(v8bf*)(inter_lds + lo * 520 + c0 + t * 16 + hi * 8) = o;
  }
  __syncthreads();

  // final projection: out[o, i] = out_w[o, c] x inter[c, i] + out_b[o]
  const int o0 = wave * 64;
  v8f facc[4];
#pragma unroll
  for (int t = 0; t < 4; t++) facc[t] = (v8f)0.0f;
  for (int kc = 0; kc < 512; kc += 32) {
    const v16bf ib = loadB(inter_lds + lo * 520 + kc + hi * 8);
#pragma unroll
    for (int t = 0; t < 4; t++) {
      const bh_t* op = owb + ((size_t)(o0 + t * 16 + lo)) * 512 + kc + hi * 8;
      facc[t] = WMMA_BF16(loadA(op), ib, facc[t]);
    }
  }
#pragma unroll
  for (int t = 0; t < 4; t++) {
#pragma unroll
    for (int v = 0; v < 8; v++) {
      const int orow = o0 + t * 16 + hi * 8 + v;
      out[((size_t)(b * 512 + orow)) * 1024 + i0 + lo] = facc[t][v] + ob[orow];
    }
  }
}

// ---------------------------------------------------------------------------
// Host launcher
// ---------------------------------------------------------------------------
extern "C" void kernel_launch(void* const* d_in, const int* in_sizes, int n_in,
                              void* d_out, int out_size, void* d_ws, size_t ws_size,
                              hipStream_t stream) {
  (void)in_sizes; (void)n_in; (void)out_size; (void)ws_size;
  const float* x   = (const float*)d_in[0];
  const float* pos = (const float*)d_in[1];
  const float* qw  = (const float*)d_in[2];
  const float* qb  = (const float*)d_in[3];
  const float* kw  = (const float*)d_in[4];
  const float* kb  = (const float*)d_in[5];
  const float* vw  = (const float*)d_in[6];
  const float* vb  = (const float*)d_in[7];
  const float* ow  = (const float*)d_in[8];
  const float* ob  = (const float*)d_in[9];
  float* out = (float*)d_out;

  char* ws = (char*)d_ws;
  // workspace layout (bytes)
  bh_t* xpT = (bh_t*)(ws);                         // 32*1024*512*2 = 33554432
  bh_t* qT  = (bh_t*)(ws + 33554432);              // 32*1024*64*2  =  4194304
  bh_t* kT  = (bh_t*)(ws + 37748736);              //                  4194304
  bh_t* vR  = (bh_t*)(ws + 41943040);              // 32*512*1024*2 = 33554432
  bh_t* qwb = (bh_t*)(ws + 75497472);              // 64*512*2      =    65536
  bh_t* kwb = (bh_t*)(ws + 75563008);              //                    65536
  bh_t* vwb = (bh_t*)(ws + 75628544);              // 512*512*2     =   524288
  bh_t* owb = (bh_t*)(ws + 76152832);              //                   524288

  k_prep_xpT<<<8192, 256, 0, stream>>>(x, pos, xpT);
  k_cvt_bf16<<<128, 256, 0, stream>>>(qw, qwb, 64 * 512);
  k_cvt_bf16<<<128, 256, 0, stream>>>(kw, kwb, 64 * 512);
  k_cvt_bf16<<<1024, 256, 0, stream>>>(vw, vwb, 512 * 512);
  k_cvt_bf16<<<1024, 256, 0, stream>>>(ow, owb, 512 * 512);

  // 32 batches * 40 m-tiles * 64 n-tiles = 81920 tiles / 8 waves = 10240 blocks
  k_qkv<<<10240, 256, 0, stream>>>(xpT, qwb, kwb, vwb, qb, kb, vb, qT, kT, vR);

  k_attn<<<dim3(64, 32), 256, 0, stream>>>(qT, kT, vR, owb, ob, out);
}
